// Net_61280593379653
// MI455X (gfx1250) — compile-verified, matrix operand order checked by reading, not run on hardware
//
#include <hip/hip_runtime.h>
#include <hip/hip_bf16.h>
#include <math.h>

typedef __attribute__((ext_vector_type(16))) __bf16 v16bf;
typedef __attribute__((ext_vector_type(8)))  float  v8f;

// ---------------------------------------------------------------------------
// Weight pack: convert Wrel/Wroot (f32 [K,32] row-major) into bf16 fragments
// laid out in exact per-lane WMMA B-operand order, so GEMM waves load each
// fragment as one contiguous 32B chunk per lane (2x global_load_b128).
//
// Layout: pack[((ks*4 + f)*32 + lane)*16 + e], f: 0=rel lo-cols, 1=rel hi,
// 2=root lo, 3=root hi.  Element e of lane's v16bf maps to
//   k = ks*32 + (lane>=16)*8 + e + (e>=8 ? 8 : 0),  n = (f&1)*16 + (lane&15)
// ---------------------------------------------------------------------------
__global__ __launch_bounds__(256)
void pack_weights(const float* __restrict__ Wrel, const float* __restrict__ Wroot,
                  __bf16* __restrict__ pack, int total /* = (K/32)*2048 */)
{
    const int tid = blockIdx.x * blockDim.x + threadIdx.x;
    if (tid >= total) return;
    const int e    = tid & 15;
    const int lane = (tid >> 4) & 31;
    const int f    = (tid >> 9) & 3;
    const int ks   = tid >> 11;
    const int laneHi = lane >> 4;
    const int lmod   = lane & 15;
    const int k = ks * 32 + laneHi * 8 + e + (e >= 8 ? 8 : 0);
    const int n = (f & 1) * 16 + lmod;
    const float* W = (f >> 1) ? Wroot : Wrel;
    pack[tid] = (__bf16)W[(size_t)k * 32 + n];
}

// ---------------------------------------------------------------------------
// Dual GEMM (WMMA bf16 -> f32):  yrel = in @ Wrel ;  acc = in @ Wroot + brel
// in: [nrows, K] f32 row-major; weights pre-packed bf16 (see pack_weights).
// One wave per 16-row tile; 4 f32 accumulators (2 matrices x 2 col halves).
// ---------------------------------------------------------------------------
__global__ __launch_bounds__(128)
void dual_gemm_wmma(const float* __restrict__ in, int K,
                    const v16bf* __restrict__ wpack,
                    const float* __restrict__ brel,
                    float* __restrict__ yrel,
                    float* __restrict__ acc,
                    int nrows)
{
    const int wave   = threadIdx.x >> 5;
    const int lane   = threadIdx.x & 31;
    const int tile   = blockIdx.x * (blockDim.x >> 5) + wave;
    const int row0   = tile * 16;
    if (row0 >= nrows) return;

    const int laneHi = lane >> 4;    // 0: lanes 0-15, 1: lanes 16-31
    const int lmod   = lane & 15;

    v8f a_rel0 = {}, a_rel1 = {}, a_rt0 = {}, a_rt1 = {};

    const int ksteps = K >> 5;
    for (int ks = 0; ks < ksteps; ++ks) {
        // ---- A tile (16x32 bf16): row = row0+lmod, two contiguous 8-float runs
        const float* arow = in + (size_t)(row0 + lmod) * K + ks * 32 + laneHi * 8;
        const float4 a0 = *(const float4*)(arow);
        const float4 a1 = *(const float4*)(arow + 4);
        const float4 a2 = *(const float4*)(arow + 16);
        const float4 a3 = *(const float4*)(arow + 20);
        v16bf a;
        a[0]  = (__bf16)a0.x; a[1]  = (__bf16)a0.y; a[2]  = (__bf16)a0.z; a[3]  = (__bf16)a0.w;
        a[4]  = (__bf16)a1.x; a[5]  = (__bf16)a1.y; a[6]  = (__bf16)a1.z; a[7]  = (__bf16)a1.w;
        a[8]  = (__bf16)a2.x; a[9]  = (__bf16)a2.y; a[10] = (__bf16)a2.z; a[11] = (__bf16)a2.w;
        a[12] = (__bf16)a3.x; a[13] = (__bf16)a3.y; a[14] = (__bf16)a3.z; a[15] = (__bf16)a3.w;

        // ---- B fragments: contiguous 32B per lane, pre-packed bf16
        const v16bf* wv = wpack + (size_t)ks * 128 + lane;
        const v16bf br0 = wv[0];
        const v16bf br1 = wv[32];
        const v16bf bo0 = wv[64];
        const v16bf bo1 = wv[96];

        a_rel0 = __builtin_amdgcn_wmma_f32_16x16x32_bf16(false, a, false, br0, (short)0, a_rel0, false, false);
        a_rel1 = __builtin_amdgcn_wmma_f32_16x16x32_bf16(false, a, false, br1, (short)0, a_rel1, false, false);
        a_rt0  = __builtin_amdgcn_wmma_f32_16x16x32_bf16(false, a, false, bo0, (short)0, a_rt0,  false, false);
        a_rt1  = __builtin_amdgcn_wmma_f32_16x16x32_bf16(false, a, false, bo1, (short)0, a_rt1,  false, false);
    }

    // ---- D store: VGPR r holds (m = r + 8*laneHi, n = lmod) and (n = 16+lmod)
    const float bia0 = brel[lmod];
    const float bia1 = brel[16 + lmod];
#pragma unroll
    for (int r = 0; r < 8; ++r) {
        const int m = r + 8 * laneHi;
        const size_t o = (size_t)(row0 + m) * 32;
        yrel[o + lmod]      = a_rel0[r];
        yrel[o + 16 + lmod] = a_rel1[r];
        acc[o + lmod]       = a_rt0[r] + bia0;
        acc[o + 16 + lmod]  = a_rt1[r] + bia1;
    }
}

// ---------------------------------------------------------------------------
// Edge scatter: acc[dst] += yrel[src] * ew. 4 threads/edge, 8 floats each.
// Features (12.8 MB) are L2-resident; atomics resolve in L2.
// ---------------------------------------------------------------------------
__global__ __launch_bounds__(256)
void edge_scatter(const int* __restrict__ src, const int* __restrict__ dst,
                  const float* __restrict__ ew,
                  const float* __restrict__ y, float* __restrict__ acc,
                  int num_edges)
{
    const int t = blockIdx.x * blockDim.x + threadIdx.x;
    const int e = t >> 2;
    if (e >= num_edges) return;
    const int c = (t & 3) * 8;
    const float w = ew[e];
    const int s = src[e];
    const int d = dst[e];
    const float4* ys = (const float4*)(y + (size_t)s * 32 + c);
    const float4 v0 = ys[0];
    const float4 v1 = ys[1];
    float* ad = acc + (size_t)d * 32 + c;
    atomicAdd(ad + 0, v0.x * w);
    atomicAdd(ad + 1, v0.y * w);
    atomicAdd(ad + 2, v0.z * w);
    atomicAdd(ad + 3, v0.w * w);
    atomicAdd(ad + 4, v1.x * w);
    atomicAdd(ad + 5, v1.y * w);
    atomicAdd(ad + 6, v1.z * w);
    atomicAdd(ad + 7, v1.w * w);
}

// ---------------------------------------------------------------------------
__global__ __launch_bounds__(256)
void relu_vec4(float4* __restrict__ a, int n4)
{
    const int t = blockIdx.x * blockDim.x + threadIdx.x;
    if (t >= n4) return;
    float4 v = a[t];
    v.x = fmaxf(v.x, 0.f); v.y = fmaxf(v.y, 0.f);
    v.z = fmaxf(v.z, 0.f); v.w = fmaxf(v.w, 0.f);
    a[t] = v;
}

__global__ __launch_bounds__(256)
void zero_f32(float* __restrict__ p, int n)
{
    const int t = blockIdx.x * blockDim.x + threadIdx.x;
    if (t < n) p[t] = 0.f;
}

// ---------------------------------------------------------------------------
// Global add pool: g[batch[n]] += h[n]. 4 threads/node, 8 floats each.
// ---------------------------------------------------------------------------
__global__ __launch_bounds__(256)
void pool_scatter(const float* __restrict__ h, const int* __restrict__ batch,
                  float* __restrict__ g, int nnodes)
{
    const int t = blockIdx.x * blockDim.x + threadIdx.x;
    const int n = t >> 2;
    if (n >= nnodes) return;
    const int c = (t & 3) * 8;
    const int b = batch[n];
    const float4* hs = (const float4*)(h + (size_t)n * 32 + c);
    const float4 v0 = hs[0];
    const float4 v1 = hs[1];
    float* gd = g + (size_t)b * 32 + c;
    atomicAdd(gd + 0, v0.x); atomicAdd(gd + 1, v0.y);
    atomicAdd(gd + 2, v0.z); atomicAdd(gd + 3, v0.w);
    atomicAdd(gd + 4, v1.x); atomicAdd(gd + 5, v1.y);
    atomicAdd(gd + 6, v1.z); atomicAdd(gd + 7, v1.w);
}

// ---------------------------------------------------------------------------
// Head: out = log_softmax(relu(g@W1 + b1) @ W2 + b2). One thread per graph.
// ---------------------------------------------------------------------------
__global__ __launch_bounds__(256)
void head_kernel(const float* __restrict__ g,
                 const float* __restrict__ W1, const float* __restrict__ b1,
                 const float* __restrict__ W2, const float* __restrict__ b2,
                 float* __restrict__ out, int ngraphs)
{
    const int i = blockIdx.x * blockDim.x + threadIdx.x;
    if (i >= ngraphs) return;
    float gv[32];
#pragma unroll
    for (int f = 0; f < 32; ++f) gv[f] = g[(size_t)i * 32 + f];
    float hd[32];
    for (int o = 0; o < 32; ++o) {
        float s = b1[o];
#pragma unroll
        for (int f = 0; f < 32; ++f) s += gv[f] * W1[f * 32 + o];
        hd[o] = fmaxf(s, 0.f);
    }
    float logits[10];
    float mx = -INFINITY;
    for (int o = 0; o < 10; ++o) {
        float s = b2[o];
#pragma unroll
        for (int f = 0; f < 32; ++f) s += hd[f] * W2[f * 10 + o];
        logits[o] = s;
        mx = fmaxf(mx, s);
    }
    float se = 0.f;
    for (int o = 0; o < 10; ++o) se += expf(logits[o] - mx);
    const float lse = mx + logf(se);
    for (int o = 0; o < 10; ++o) out[(size_t)i * 10 + o] = logits[o] - lse;
}

// ---------------------------------------------------------------------------
extern "C" void kernel_launch(void* const* d_in, const int* in_sizes, int n_in,
                              void* d_out, int out_size, void* d_ws, size_t ws_size,
                              hipStream_t stream)
{
    const float* x     = (const float*)d_in[0];
    const int*   ei    = (const int*)d_in[1];
    const int*   batch = (const int*)d_in[2];
    const float* ew    = (const float*)d_in[3];

    const float* Wrel[5], *brel[5], *Wroot[5];
    for (int l = 0; l < 5; ++l) {
        Wrel[l]  = (const float*)d_in[4 + 3 * l];
        brel[l]  = (const float*)d_in[5 + 3 * l];
        Wroot[l] = (const float*)d_in[6 + 3 * l];
    }
    const float* Wlin1 = (const float*)d_in[19];
    const float* blin1 = (const float*)d_in[20];
    const float* Wlin2 = (const float*)d_in[21];
    const float* blin2 = (const float*)d_in[22];
    float* out = (float*)d_out;

    const int N = in_sizes[2];            // 100000 nodes
    const int E = in_sizes[1] / 2;        // 1600000 edges
    const int F = in_sizes[0] / N;        // 128 features
    const int G = out_size / 10;          // 1000 graphs

    const int* src = ei;
    const int* dst = ei + E;

    // Workspace layout (all regions 256B-aligned)
    char* ws = (char*)d_ws;
    const size_t featBytes = (size_t)N * 32 * sizeof(float);
    const size_t gBytes    = (((size_t)G * 32 * sizeof(float)) + 255) & ~(size_t)255;
    float*  bufY  = (float*)ws;                              // yrel scratch
    float*  bufA  = (float*)(ws + featBytes);                // h / acc ping
    float*  bufB  = (float*)(ws + 2 * featBytes);            // h / acc pong
    float*  gbuf  = (float*)(ws + 3 * featBytes);            // [G,32] pooled
    __bf16* wpack = (__bf16*)(ws + 3 * featBytes + gBytes);  // packed weights

    const int tiles      = (N + 15) / 16;
    const dim3 gemmGrid((tiles + 3) / 4);
    const dim3 gemmBlk(128);
    const int edgeBlocks = (4 * E + 255) / 256;
    const int reluBlocks = (N * 8 + 255) / 256;              // N*32/4 float4s
    const int poolBlocks = (4 * N + 255) / 256;

    const float* layer_in = x;
    float* accs[5] = { bufA, bufB, bufA, bufB, bufA };
    int Kdim = F;

    for (int l = 0; l < 5; ++l) {
        float* acc = accs[l];
        // pack weights into WMMA fragment order (bf16)
        const int packTotal = (Kdim >> 5) * 2048;
        pack_weights<<<(packTotal + 255) / 256, 256, 0, stream>>>(
            Wrel[l], Wroot[l], wpack, packTotal);
        // yrel = h@Wrel ; acc = h@Wroot + brel   (WMMA)
        dual_gemm_wmma<<<gemmGrid, gemmBlk, 0, stream>>>(
            layer_in, Kdim, (const v16bf*)wpack, brel[l], bufY, acc, N);
        // acc[dst] += yrel[src] * ew
        edge_scatter<<<edgeBlocks, 256, 0, stream>>>(src, dst, ew, bufY, acc, E);
        // h = relu(acc)
        relu_vec4<<<reluBlocks, 256, 0, stream>>>((float4*)acc, N * 8);
        layer_in = acc;
        Kdim = 32;
    }

    // global add pool
    zero_f32<<<(G * 32 + 255) / 256, 256, 0, stream>>>(gbuf, G * 32);
    pool_scatter<<<poolBlocks, 256, 0, stream>>>(layer_in, batch, gbuf, N);

    // MLP head + log_softmax
    head_kernel<<<(G + 255) / 256, 256, 0, stream>>>(
        gbuf, Wlin1, blin1, Wlin2, blin2, out, G);
}
